// HybridKernelMethod_65481071400658
// MI455X (gfx1250) — matrix operand order repaired and checked by reading, not compiled, not run on hardware
//
#include <hip/hip_runtime.h>
#include <hip/hip_bf16.h>

typedef __attribute__((ext_vector_type(16))) _Float16 v16h;
typedef __attribute__((ext_vector_type(8)))  _Float16 v8h;
typedef __attribute__((ext_vector_type(8)))  float    v8f;
typedef __attribute__((ext_vector_type(4)))  int      v4i;

#define N_IMG   4096
#define KDIM    784
#define KPAD    800          // 25 * 32
#define KSTEPS  25           // KPAD / 32
#define LDS_STR 40           // padded row stride (halves): 80B -> conflict-free for 16-lane b128

// ---- CDNA5 async global->LDS path (guarded; fallback = sync staging) -------
#if defined(__has_builtin)
#if __has_builtin(__builtin_amdgcn_global_load_async_to_lds_b128)
#define HAVE_ASYNC_LDS 1
#endif
#endif

#if defined(HAVE_ASYNC_LDS)
typedef __attribute__((address_space(1))) v4i gv4i_t;   // global int4
typedef __attribute__((address_space(3))) v4i lv4i_t;   // LDS int4
__device__ __forceinline__ void async_cp_b128(const void* g, void* l) {
    __builtin_amdgcn_global_load_async_to_lds_b128(
        (gv4i_t*)g, (lv4i_t*)l, 0, 0);
}
__device__ __forceinline__ void wait_async0() {
#if __has_builtin(__builtin_amdgcn_s_wait_asynccnt)
    __builtin_amdgcn_s_wait_asynccnt(0);
#else
    asm volatile("s_wait_asynccnt 0" ::: "memory");
#endif
}
#endif

// ---------------------------------------------------------------------------
// Kernel 1: CNN feature extractor. One block (256 threads) per image.
// conv1(1->8,3x3,SAME)+relu+pool2 ; conv2(8->16,3x3,SAME)+relu+pool2
// -> 784 f16 features (zero-padded to 800) + fp32 squared norm.
// ---------------------------------------------------------------------------
__global__ __launch_bounds__(256)
void extract_kernel(const float* __restrict__ in,
                    const float* __restrict__ w1, const float* __restrict__ b1,
                    const float* __restrict__ w2, const float* __restrict__ b2,
                    _Float16* __restrict__ feat, float* __restrict__ norm)
{
    __shared__ float s_img[28 * 28];
    __shared__ float s_w1[72];
    __shared__ float s_b1[8];
    __shared__ float s_w2[1152];
    __shared__ float s_b2[16];
    __shared__ float s_pool1[8 * 14 * 14];
    __shared__ float s_red[256];

    const int t   = threadIdx.x;
    const int img = blockIdx.x;

    for (int i = t; i < 784; i += 256)  s_img[i] = in[img * 784 + i];
    for (int i = t; i < 72;  i += 256)  s_w1[i]  = w1[i];
    for (int i = t; i < 1152; i += 256) s_w2[i]  = w2[i];
    if (t < 8)  s_b1[t] = b1[t];
    if (t < 16) s_b2[t] = b2[t];
    __syncthreads();

    // stage 1: relu(conv1)+maxpool -> s_pool1[8][14][14]
    for (int idx = t; idx < 8 * 14 * 14; idx += 256) {
        const int c = idx / 196, rem = idx % 196;
        const int pi = rem / 14, pj = rem % 14;
        float m = 0.0f;                      // relu makes outputs >= 0
        for (int py = 0; py < 2; ++py)
        for (int px = 0; px < 2; ++px) {
            const int h = 2 * pi + py, w = 2 * pj + px;
            float acc = s_b1[c];
            #pragma unroll
            for (int dy = 0; dy < 3; ++dy) {
                const int y = h + dy - 1;
                if (y < 0 || y > 27) continue;
                #pragma unroll
                for (int dx = 0; dx < 3; ++dx) {
                    const int x = w + dx - 1;
                    if (x < 0 || x > 27) continue;
                    acc = __builtin_fmaf(s_img[y * 28 + x], s_w1[c * 9 + dy * 3 + dx], acc);
                }
            }
            m = fmaxf(m, fmaxf(acc, 0.0f));
        }
        s_pool1[idx] = m;
    }
    __syncthreads();

    // stage 2: relu(conv2)+maxpool -> 784 features, f16, + squared norm
    float sq = 0.0f;
    for (int f = t; f < KPAD; f += 256) {
        if (f < KDIM) {
            const int c = f / 49, rem = f % 49;
            const int pi = rem / 7, pj = rem % 7;
            float m = 0.0f;
            for (int py = 0; py < 2; ++py)
            for (int px = 0; px < 2; ++px) {
                const int h = 2 * pi + py, w = 2 * pj + px;
                float acc = s_b2[c];
                #pragma unroll
                for (int cc = 0; cc < 8; ++cc) {
                    const float* wp = &s_w2[(c * 8 + cc) * 9];
                    const float* pp = &s_pool1[cc * 196];
                    #pragma unroll
                    for (int dy = 0; dy < 3; ++dy) {
                        const int y = h + dy - 1;
                        if (y < 0 || y > 13) continue;
                        #pragma unroll
                        for (int dx = 0; dx < 3; ++dx) {
                            const int x = w + dx - 1;
                            if (x < 0 || x > 13) continue;
                            acc = __builtin_fmaf(pp[y * 14 + x], wp[dy * 3 + dx], acc);
                        }
                    }
                }
                m = fmaxf(m, fmaxf(acc, 0.0f));
            }
            feat[(size_t)img * KPAD + f] = (_Float16)m;
            sq = __builtin_fmaf(m, m, sq);
        } else {
            feat[(size_t)img * KPAD + f] = (_Float16)0.0f;  // K padding
        }
    }

    s_red[t] = sq;
    __syncthreads();
    for (int off = 128; off > 0; off >>= 1) {
        if (t < off) s_red[t] += s_red[t + off];
        __syncthreads();
    }
    if (t == 0) norm[img] = s_red[0];
}

// ---------------------------------------------------------------------------
// Kernel 2: RBF kernel matrix via WMMA GEMM.
// Block: 256 threads (8 waves), 128x128 output tile.
// Wave (2x4 grid) owns 64x32 = 4x2 tiles of 16x16; K-loop 25 steps of 32.
// Async path: double-buffered LDS fed by GLOBAL_LOAD_ASYNC_TO_LDS_B128,
// synchronized with s_wait_asynccnt + workgroup barrier.
// ---------------------------------------------------------------------------
__global__ __launch_bounds__(256)
void rbf_wmma_kernel(const _Float16* __restrict__ fx,
                     const _Float16* __restrict__ fy,
                     const float* __restrict__ nx,
                     const float* __restrict__ ny,
                     float* __restrict__ out)
{
#if defined(HAVE_ASYNC_LDS)
    __shared__ __align__(16) _Float16 As[2][128 * LDS_STR];  // [m][k] tiles
    __shared__ __align__(16) _Float16 Bs[2][128 * LDS_STR];  // [n][k] tiles (B^T)
#else
    __shared__ __align__(16) _Float16 As[1][128 * LDS_STR];
    __shared__ __align__(16) _Float16 Bs[1][128 * LDS_STR];
#endif

    const int t     = threadIdx.x;
    const int lane  = t & 31;
    const int wave  = t >> 5;
    const int waveM = wave >> 2;       // 0..1 -> 64-row slab
    const int waveN = wave & 3;        // 0..3 -> 32-col slab
    const int laneM = lane & 15;
    const int hi    = lane >> 4;       // lane half

    const int blockM = blockIdx.y * 128;
    const int blockN = blockIdx.x * 128;

    // global staging coords: each thread moves one 16-half row-chunk per tile
    const int lrow  = t >> 1;          // 0..127
    const int lhalf = t & 1;           // 0..1 (16-half chunk)

    const _Float16* gA0 = fx + (size_t)(blockM + lrow) * KPAD + lhalf * 16;
    const _Float16* gB0 = fy + (size_t)(blockN + lrow) * KPAD + lhalf * 16;
    const int ldsOff = lrow * LDS_STR + lhalf * 16;

    v8f acc[4][2];
    #pragma unroll
    for (int mr = 0; mr < 4; ++mr)
        #pragma unroll
        for (int nc = 0; nc < 2; ++nc)
            acc[mr][nc] = (v8f){0.f, 0.f, 0.f, 0.f, 0.f, 0.f, 0.f, 0.f};

#if defined(HAVE_ASYNC_LDS)
    // prologue: async-stage tile 0 into buffer 0
    {
        async_cp_b128(gA0,     &As[0][ldsOff]);
        async_cp_b128(gA0 + 8, &As[0][ldsOff + 8]);
        async_cp_b128(gB0,     &Bs[0][ldsOff]);
        async_cp_b128(gB0 + 8, &Bs[0][ldsOff + 8]);
    }
    int cur = 0;
#endif

    for (int ks = 0; ks < KSTEPS; ++ks) {
#if defined(HAVE_ASYNC_LDS)
        wait_async0();                 // this wave's async fills of buf[cur] done
        __syncthreads();               // all waves done filling (and done reading buf[cur^1])
        if (ks + 1 < KSTEPS) {         // prefetch next tile into the other buffer
            const int ko = (ks + 1) * 32;
            async_cp_b128(gA0 + ko,     &As[cur ^ 1][ldsOff]);
            async_cp_b128(gA0 + ko + 8, &As[cur ^ 1][ldsOff + 8]);
            async_cp_b128(gB0 + ko,     &Bs[cur ^ 1][ldsOff]);
            async_cp_b128(gB0 + ko + 8, &Bs[cur ^ 1][ldsOff + 8]);
        }
        const _Float16* Ab = As[cur];
        const _Float16* Bb = Bs[cur];
#else
        // sync fallback: VGPR round-trip staging
        {
            const int ko = ks * 32;
            v8h a0 = *(const v8h*)(gA0 + ko);
            v8h a1 = *(const v8h*)(gA0 + ko + 8);
            v8h b0 = *(const v8h*)(gB0 + ko);
            v8h b1 = *(const v8h*)(gB0 + ko + 8);
            __syncthreads();           // protect previous iteration's reads
            *(v8h*)(&As[0][ldsOff])     = a0;
            *(v8h*)(&As[0][ldsOff + 8]) = a1;
            *(v8h*)(&Bs[0][ldsOff])     = b0;
            *(v8h*)(&Bs[0][ldsOff + 8]) = b1;
        }
        __syncthreads();
        const _Float16* Ab = As[0];
        const _Float16* Bb = Bs[0];
#endif

        // B fragments: lane holds column N=laneM, K = hi*16 .. hi*16+15 (contiguous)
        v16h bf[2];
        #pragma unroll
        for (int nc = 0; nc < 2; ++nc) {
            const _Float16* p = &Bb[(waveN * 32 + nc * 16 + laneM) * LDS_STR + hi * 16];
            v8h lo  = *(const v8h*)(p);
            v8h hi8 = *(const v8h*)(p + 8);
            bf[nc] = __builtin_shufflevector(lo, hi8, 0,1,2,3,4,5,6,7,8,9,10,11,12,13,14,15);
        }
        // A fragments: lane holds row M=laneM, K = hi*16 .. hi*16+15 (contiguous)
        #pragma unroll
        for (int mr = 0; mr < 4; ++mr) {
            const _Float16* p = &Ab[(waveM * 64 + mr * 16 + laneM) * LDS_STR];
            v8h lo  = *(const v8h*)(p + hi * 16);
            v8h hi8 = *(const v8h*)(p + hi * 16 + 8);
            v16h af = __builtin_shufflevector(lo, hi8, 0,1,2,3,4,5,6,7,8,9,10,11,12,13,14,15);
            #pragma unroll
            for (int nc = 0; nc < 2; ++nc) {
                acc[mr][nc] = __builtin_amdgcn_wmma_f32_16x16x32_f16(
                    false, af, false, bf[nc], (short)0, acc[mr][nc], false, false);
            }
        }
#if defined(HAVE_ASYNC_LDS)
        cur ^= 1;
#endif
    }

    // epilogue: K[m,n] = exp(-max(||x||^2 + ||y||^2 - 2*xy, 0))
    #pragma unroll
    for (int mr = 0; mr < 4; ++mr) {
        const int Mbase = blockM + waveM * 64 + mr * 16 + 8 * hi;  // + r for VGPR r
        float xn[8];
        #pragma unroll
        for (int r = 0; r < 8; ++r) xn[r] = nx[Mbase + r];
        #pragma unroll
        for (int nc = 0; nc < 2; ++nc) {
            const int Ncol = blockN + waveN * 32 + nc * 16 + laneM;
            const float yn = ny[Ncol];
            #pragma unroll
            for (int r = 0; r < 8; ++r) {
                float sq = fmaxf(xn[r] + yn - 2.0f * acc[mr][nc][r], 0.0f);
                out[(size_t)(Mbase + r) * N_IMG + Ncol] = __expf(-sq);
            }
        }
    }
}

// ---------------------------------------------------------------------------
extern "C" void kernel_launch(void* const* d_in, const int* in_sizes, int n_in,
                              void* d_out, int out_size, void* d_ws, size_t ws_size,
                              hipStream_t stream)
{
    const float* x  = (const float*)d_in[0];
    const float* y  = (const float*)d_in[1];
    const float* w1 = (const float*)d_in[2];
    const float* b1 = (const float*)d_in[3];
    const float* w2 = (const float*)d_in[4];
    const float* b2 = (const float*)d_in[5];
    float* out = (float*)d_out;

    const size_t featBytes = (size_t)N_IMG * KPAD * sizeof(_Float16);  // 6.25 MB
    char* ws = (char*)d_ws;
    _Float16* fx = (_Float16*)(ws);
    _Float16* fy = (_Float16*)(ws + featBytes);
    float*    nx = (float*)(ws + 2 * featBytes);
    float*    ny = nx + N_IMG;

    extract_kernel<<<N_IMG, 256, 0, stream>>>(x, w1, b1, w2, b2, fx, nx);
    extract_kernel<<<N_IMG, 256, 0, stream>>>(y, w1, b1, w2, b2, fy, ny);

    dim3 grid(N_IMG / 128, N_IMG / 128);   // 32 x 32 blocks
    rbf_wmma_kernel<<<grid, 256, 0, stream>>>(fx, fy, nx, ny, out);
}
